// CausalAttention_74594991997127
// MI455X (gfx1250) — matrix operand order repaired and checked by reading, not compiled, not run on hardware
//
#include <hip/hip_runtime.h>
#include <hip/hip_bf16.h>
#include <cstdint>
#include <cstddef>

typedef __bf16 bf16;
typedef __attribute__((ext_vector_type(16))) __bf16 bf16x16;
typedef __attribute__((ext_vector_type(8)))  float   floatx8;

#define B_SZ    2
#define T_SEQ   2048
#define D_MODEL 2048
#define N_QH    16
#define N_KVH   4
#define HEAD_D  128
#define SM_SCALE 0.08838834764831845f   // 1/sqrt(128)

// CDNA5 async global->LDS copy path (ASYNCcnt-tracked, bypasses VGPRs).
#define USE_ASYNC_G2L 1

union FragU { bf16x16 v; uint4 q[2]; };

// A-fragment (16x32 bf16): lane m=l16; half=0 -> elems K[0..7],K[16..23]; half=1 -> K[8..15],K[24..31]
__device__ __forceinline__ bf16x16 frag_a(const bf16* tile, int mBase, int kBase,
                                          int ld, int l16, int hl) {
  FragU f;
  const bf16* p = tile + (size_t)(mBase + l16) * ld + kBase + hl * 8;
  f.q[0] = *(const uint4*)(p);
  f.q[1] = *(const uint4*)(p + 16);
  return f.v;
}

// B-fragment (32x16 bf16) from an N-major (N x K) tile: lane n=l16; half=0 -> elems K[0..7],K[16..23]; half=1 -> K[8..15],K[24..31]
__device__ __forceinline__ bf16x16 frag_b(const bf16* tile, int nBase, int kBase,
                                          int ld, int l16, int hl) {
  FragU f;
  const bf16* p = tile + (size_t)(nBase + l16) * ld + kBase + hl * 8;
  f.q[0] = *(const uint4*)(p);
  f.q[1] = *(const uint4*)(p + 16);
  return f.v;
}

// 16B global -> LDS copy. Async form: VDST = per-lane LDS byte offset (low 32
// bits of the flat pointer into __shared__ are the LDS offset per the CDNA5
// aperture rules), VADDR = 64-bit global address.
__device__ __forceinline__ void copy16_g2l(bf16* ldst, const bf16* gsrc) {
#if USE_ASYNC_G2L
  const uint32_t loff = (uint32_t)(uintptr_t)ldst;
  asm volatile("global_load_async_to_lds_b128 %0, %1, off"
               :: "v"(loff), "v"(gsrc) : "memory");
#else
  *(uint4*)ldst = *(const uint4*)gsrc;
#endif
}

__device__ __forceinline__ void async_wait() {
#if USE_ASYNC_G2L
  asm volatile("s_wait_asynccnt 0x0" ::: "memory");
#endif
}

// ---------------------------------------------------------------- prep kernels

__global__ __launch_bounds__(256)
void f32_to_bf16_vec(const float* __restrict__ src, bf16* __restrict__ dst, int n) {
  const int i = (blockIdx.x * 256 + threadIdx.x) * 4;
  if (i >= n) return;
  const float4 f = *(const float4*)(src + i);
  dst[i + 0] = (bf16)f.x;
  dst[i + 1] = (bf16)f.y;
  dst[i + 2] = (bf16)f.z;
  dst[i + 3] = (bf16)f.w;
}

// src: K x N (f32, row-major) -> dst: N x K (bf16, row-major)
__global__ __launch_bounds__(256)
void transpose_f32_to_bf16(const float* __restrict__ src, bf16* __restrict__ dst,
                           int K, int N) {
  __shared__ float tile[32][33];
  const int nb = blockIdx.x * 32, kb = blockIdx.y * 32;
  const int tx = threadIdx.x & 31;
  const int ty = threadIdx.x >> 5;   // 0..7
#pragma unroll
  for (int i = 0; i < 4; ++i)
    tile[ty + i * 8][tx] = src[(size_t)(kb + ty + i * 8) * N + nb + tx];
  __syncthreads();
#pragma unroll
  for (int i = 0; i < 4; ++i)
    dst[(size_t)(nb + ty + i * 8) * K + kb + tx] = (bf16)tile[tx][ty + i * 8];
}

// -------------------------------------------------------------------- WMMA GEMM
// C(MxN) = A(MxK, bf16 row-major) * Bt(NxK, bf16 row-major)^T
// MODE 0: f32 row-major output. MODE 1: bf16 output scattered to (B, H, T, HD).
template<int MODE>
__global__ __launch_bounds__(256)
void gemm128(const bf16* __restrict__ A, const bf16* __restrict__ Bt,
             float* __restrict__ outF, bf16* __restrict__ outB,
             int M, int N, int K, int H) {
  constexpr int LD = 72;   // 144B row stride -> conflict-free b128 fragment loads
  __shared__ bf16 As[128 * LD];    // 128 x 64-K slice (stride-padded)
  __shared__ bf16 Bs[128 * LD];
  const int tid = threadIdx.x;
  const int w = tid >> 5, lane = tid & 31, hl = lane >> 4, l16 = lane & 15;
  const int bRow = blockIdx.y * 128, bCol = blockIdx.x * 128;
  const int wr = (w >> 1) * 32, wc = (w & 1) * 64;
  const floatx8 z8 = {0.f,0.f,0.f,0.f,0.f,0.f,0.f,0.f};
  floatx8 acc[2][4];
#pragma unroll
  for (int i = 0; i < 2; ++i)
#pragma unroll
    for (int j = 0; j < 4; ++j) acc[i][j] = z8;

  for (int k0 = 0; k0 < K; k0 += 64) {
    __syncthreads();
#pragma unroll
    for (int i = 0; i < 4; ++i) {
      const int c = tid + i * 256;          // 1024 chunks of 16B per tile
      const int r = c >> 3, off = (c & 7) * 8;
      copy16_g2l(&As[r * LD + off], A  + (size_t)(bRow + r) * K + k0 + off);
      copy16_g2l(&Bs[r * LD + off], Bt + (size_t)(bCol + r) * K + k0 + off);
      if (k0 + 64 < K) {
        __builtin_prefetch(A  + (size_t)(bRow + r) * K + k0 + 64 + off, 0, 3);
        __builtin_prefetch(Bt + (size_t)(bCol + r) * K + k0 + 64 + off, 0, 3);
      }
    }
    async_wait();
    __syncthreads();
#pragma unroll
    for (int ks = 0; ks < 2; ++ks) {
      bf16x16 af[2];
#pragma unroll
      for (int mi = 0; mi < 2; ++mi)
        af[mi] = frag_a(As, wr + mi * 16, ks * 32, LD, l16, hl);
#pragma unroll
      for (int nj = 0; nj < 4; ++nj) {
        const bf16x16 bb = frag_b(Bs, wc + nj * 16, ks * 32, LD, l16, hl);
#pragma unroll
        for (int mi = 0; mi < 2; ++mi)
          acc[mi][nj] = __builtin_amdgcn_wmma_f32_16x16x32_bf16(
              false, af[mi], false, bb, (short)0, acc[mi][nj], false, false);
      }
    }
  }

#pragma unroll
  for (int mi = 0; mi < 2; ++mi)
#pragma unroll
    for (int nj = 0; nj < 4; ++nj)
#pragma unroll
      for (int v = 0; v < 8; ++v) {
        const float val = acc[mi][nj][v];
        const int rg = bRow + wr + mi * 16 + hl * 8 + v;   // C layout: row = v + 8*half
        const int cg = bCol + wc + nj * 16 + l16;          // col = lane&15
        if (MODE == 0) {
          outF[(size_t)rg * N + cg] = val;
        } else {
          const int bb2 = rg / T_SEQ, t = rg % T_SEQ;
          const int hh = cg / HEAD_D, hd = cg % HEAD_D;
          outB[(((size_t)bb2 * H + hh) * T_SEQ + t) * HEAD_D + hd] = (bf16)val;
        }
      }
}

// ----------------------------------------------------------------------- RoPE
// In-place RoPE on bf16 Q (B,QH,T,HD) and K (B,KVH,T,HD). One thread per (row, d<64) pair.
__global__ __launch_bounds__(256)
void rope_kernel(bf16* __restrict__ Qb, bf16* __restrict__ Kb) {
  const int NQ = B_SZ * N_QH  * T_SEQ * (HEAD_D / 2);
  const int NK = B_SZ * N_KVH * T_SEQ * (HEAD_D / 2);
  const int idx = blockIdx.x * 256 + threadIdx.x;
  if (idx >= NQ + NK) return;
  bf16* p; int i;
  if (idx < NQ) { p = Qb; i = idx; } else { p = Kb; i = idx - NQ; }
  const int d  = i & 63;
  const int t  = (i >> 6) & (T_SEQ - 1);
  const int bh = i >> 17;                 // 6 (d) + 11 (t) bits
  const size_t base = ((size_t)bh * T_SEQ + t) * HEAD_D;
  const float x1 = (float)p[base + d];
  const float x2 = (float)p[base + d + 64];
  // freq = 10000^(-2d/HD) = exp(-2d * ln(10000)/HD)
  const float freq = __expf(-(float)(2 * d) * (9.210340371976184f / (float)HEAD_D));
  const float ang = (float)t * freq;
  const float c = cosf(ang), s = sinf(ang);
  p[base + d]      = (bf16)(x1 * c - x2 * s);
  p[base + d + 64] = (bf16)(x2 * c + x1 * s);
}

// ----------------------------------------------------------- flash attention
// Grid: (T/128 qtiles, QH, B). 256 threads = 8 waves; wave w owns query rows 16w..16w+15.
__global__ __launch_bounds__(256)
void attn_kernel(const bf16* __restrict__ Qb, const bf16* __restrict__ Kb,
                 const bf16* __restrict__ Vb, bf16* __restrict__ attnOut) {
  constexpr int LDK  = 136;  // key-tile row stride (128 hd + 8 pad), 272B
  constexpr int LDP  = 72;   // Vt / P row stride (64 keys + 8 pad), 144B
  __shared__ bf16 Ks [64  * LDK];   // keys x hd
  __shared__ bf16 Vts[128 * LDP];   // hd x keys (transposed)
  __shared__ bf16 Ps [128 * LDP];   // queries x keys (per-wave rows)

  const int qb = blockIdx.x, h = blockIdx.y, b = blockIdx.z;
  const int kvh = h >> 2;                      // GQA: 4 q-heads per kv-head
  const int tid = threadIdx.x;
  const int w = tid >> 5, lane = tid & 31, hl = lane >> 4, l16 = lane & 15;
  const int q0 = qb * 128;

  const bf16* Qh = Qb + ((size_t)(b * N_QH  + h  ) * T_SEQ) * HEAD_D;
  const bf16* Kh = Kb + ((size_t)(b * N_KVH + kvh) * T_SEQ) * HEAD_D;
  const bf16* Vh = Vb + ((size_t)(b * N_KVH + kvh) * T_SEQ) * HEAD_D;

  // Q fragments stay in registers: 4 K-steps of 32 across head_dim=128
  bf16x16 qf[4];
  {
    const bf16* qrow = Qh + (size_t)(q0 + 16 * w + l16) * HEAD_D;
#pragma unroll
    for (int ks = 0; ks < 4; ++ks) {
      FragU f;
      f.q[0] = *(const uint4*)(qrow + ks * 32 + hl * 8);
      f.q[1] = *(const uint4*)(qrow + ks * 32 + 16 + hl * 8);
      qf[ks] = f.v;
    }
  }

  const floatx8 z8 = {0.f,0.f,0.f,0.f,0.f,0.f,0.f,0.f};
  floatx8 o[8];
  float m_run[8], l_run[8];
#pragma unroll
  for (int i = 0; i < 8; ++i) { o[i] = z8; m_run[i] = -1e30f; l_run[i] = 0.f; }

  const int qg0 = q0 + 16 * w + hl * 8;   // query row for acc element v is qg0 + v
  const int nkt = 2 * qb + 2;             // 64-key tiles covering keys <= q0+127
  for (int j = 0; j < nkt; ++j) {
    const int kg0 = j * 64;
    __syncthreads();
    // cooperative load: K tile async (row-major) + V tile transposed into Vts
    for (int c = tid; c < 64 * 16; c += 256) {
      const int r = c >> 4, off = (c & 15) * 8;
      copy16_g2l(&Ks[r * LDK + off], Kh + (size_t)(kg0 + r) * HEAD_D + off);
      const uint4 tv = *(const uint4*)(Vh + (size_t)(kg0 + r) * HEAD_D + off);
      const bf16* e = (const bf16*)&tv;
#pragma unroll
      for (int i = 0; i < 8; ++i) Vts[(off + i) * LDP + r] = e[i];
    }
    async_wait();
    __syncthreads();

    // S = Q K^T  (16 queries x 64 keys per wave)
    floatx8 s[4];
#pragma unroll
    for (int i = 0; i < 4; ++i) s[i] = z8;
#pragma unroll
    for (int ks = 0; ks < 4; ++ks)
#pragma unroll
      for (int tc = 0; tc < 4; ++tc) {
        const bf16x16 kf = frag_b(Ks, tc * 16, ks * 32, LDK, l16, hl);
        s[tc] = __builtin_amdgcn_wmma_f32_16x16x32_bf16(
            false, qf[ks], false, kf, (short)0, s[tc], false, false);
      }

    // scale + causal mask
#pragma unroll
    for (int tc = 0; tc < 4; ++tc)
#pragma unroll
      for (int v = 0; v < 8; ++v) {
        float val = s[tc][v] * SM_SCALE;
        const int key = kg0 + tc * 16 + l16;
        if (key > qg0 + v) val = -1e30f;
        s[tc][v] = val;
      }

    // online softmax (per-row reduce across the 16 lanes of each half)
#pragma unroll
    for (int v = 0; v < 8; ++v) {
      float mx = fmaxf(fmaxf(s[0][v], s[1][v]), fmaxf(s[2][v], s[3][v]));
#pragma unroll
      for (int msk = 1; msk <= 8; msk <<= 1) mx = fmaxf(mx, __shfl_xor(mx, msk, 32));
      const float mn = fmaxf(m_run[v], mx);
      const float sc = __expf(m_run[v] - mn);
      m_run[v] = mn;
      float rs = 0.f;
#pragma unroll
      for (int tc = 0; tc < 4; ++tc) {
        const float pp = __expf(s[tc][v] - mn);
        s[tc][v] = pp;
        rs += pp;
      }
#pragma unroll
      for (int msk = 1; msk <= 8; msk <<= 1) rs += __shfl_xor(rs, msk, 32);
      l_run[v] = l_run[v] * sc + rs;
#pragma unroll
      for (int tc = 0; tc < 8; ++tc) o[tc][v] *= sc;
    }

    // stage P (C-layout -> A-fragment layout) through per-wave LDS rows
#pragma unroll
    for (int tc = 0; tc < 4; ++tc)
#pragma unroll
      for (int v = 0; v < 8; ++v)
        Ps[(size_t)(16 * w + hl * 8 + v) * LDP + tc * 16 + l16] = (bf16)s[tc][v];

    // O += P @ V   (K-dim = 64 keys -> 2 steps of 32)
#pragma unroll
    for (int ks = 0; ks < 2; ++ks) {
      const bf16x16 pf = frag_a(Ps, 16 * w, ks * 32, LDP, l16, hl);
#pragma unroll
      for (int tc = 0; tc < 8; ++tc) {
        const bf16x16 vf = frag_b(Vts, tc * 16, ks * 32, LDP, l16, hl);
        o[tc] = __builtin_amdgcn_wmma_f32_16x16x32_bf16(
            false, pf, false, vf, (short)0, o[tc], false, false);
      }
    }
  }

  // normalize + store: attn_out (B, T, D) with col = h*128 + hd
#pragma unroll
  for (int tc = 0; tc < 8; ++tc)
#pragma unroll
    for (int v = 0; v < 8; ++v) {
      const float val = o[tc][v] / l_run[v];
      const int tg  = qg0 + v;
      const int col = h * HEAD_D + tc * 16 + l16;
      attnOut[((size_t)(b * T_SEQ + tg)) * D_MODEL + col] = (bf16)val;
    }
}

// ------------------------------------------------------------------- launcher

extern "C" void kernel_launch(void* const* d_in, const int* in_sizes, int n_in,
                              void* d_out, int out_size, void* d_ws, size_t ws_size,
                              hipStream_t stream) {
  (void)in_sizes; (void)n_in; (void)out_size; (void)ws_size;
  const float* x  = (const float*)d_in[0];
  const float* wq = (const float*)d_in[1];
  const float* wk = (const float*)d_in[2];
  const float* wv = (const float*)d_in[3];
  const float* wo = (const float*)d_in[4];
  float* out = (float*)d_out;

  char* ws = (char*)d_ws;
  size_t off = 0;
  auto take = [&](size_t bytes) -> char* {
    char* p = ws + off;
    off += (bytes + 255) & ~(size_t)255;
    return p;
  };

  const size_t MT  = (size_t)B_SZ * T_SEQ;        // 4096 rows
  const int    NKV = N_KVH * HEAD_D;              // 512

  bf16* xb   = (bf16*)take(MT * D_MODEL * 2);                  // x in bf16
  bf16* wqt  = (bf16*)take((size_t)D_MODEL * D_MODEL * 2);     // wq^T
  bf16* wkt  = (bf16*)take((size_t)NKV * D_MODEL * 2);         // wk^T
  bf16* wvt  = (bf16*)take((size_t)NKV * D_MODEL * 2);         // wv^T
  bf16* wot  = (bf16*)take((size_t)D_MODEL * D_MODEL * 2);     // wo^T
  bf16* Qb   = (bf16*)take(MT * D_MODEL * 2);                  // (B,QH,T,HD)
  bf16* Kb   = (bf16*)take(MT * NKV * 2);                      // (B,KVH,T,HD)
  bf16* Vb   = (bf16*)take(MT * NKV * 2);                      // (B,KVH,T,HD)
  bf16* attn = (bf16*)take(MT * D_MODEL * 2);                  // (B,T,D)

  // 1) precision staging
  const int nx = B_SZ * T_SEQ * D_MODEL;
  f32_to_bf16_vec<<<nx / 4 / 256, 256, 0, stream>>>(x, xb, nx);
  transpose_f32_to_bf16<<<dim3(D_MODEL / 32, D_MODEL / 32), 256, 0, stream>>>(wq, wqt, D_MODEL, D_MODEL);
  transpose_f32_to_bf16<<<dim3(NKV / 32,     D_MODEL / 32), 256, 0, stream>>>(wk, wkt, D_MODEL, NKV);
  transpose_f32_to_bf16<<<dim3(NKV / 32,     D_MODEL / 32), 256, 0, stream>>>(wv, wvt, D_MODEL, NKV);
  transpose_f32_to_bf16<<<dim3(D_MODEL / 32, D_MODEL / 32), 256, 0, stream>>>(wo, wot, D_MODEL, D_MODEL);

  // 2) QKV projections (epilogue scatters into head-major bf16 layouts)
  gemm128<1><<<dim3(D_MODEL / 128, (int)MT / 128), 256, 0, stream>>>(
      xb, wqt, nullptr, Qb, (int)MT, D_MODEL, D_MODEL, N_QH);
  gemm128<1><<<dim3(NKV / 128, (int)MT / 128), 256, 0, stream>>>(
      xb, wkt, nullptr, Kb, (int)MT, NKV, D_MODEL, N_KVH);
  gemm128<1><<<dim3(NKV / 128, (int)MT / 128), 256, 0, stream>>>(
      xb, wvt, nullptr, Vb, (int)MT, NKV, D_MODEL, N_KVH);

  // 3) RoPE on Q and K (in place)
  const int nrope = B_SZ * (N_QH + N_KVH) * T_SEQ * (HEAD_D / 2);
  rope_kernel<<<(nrope + 255) / 256, 256, 0, stream>>>(Qb, Kb);

  // 4) causal flash attention
  attn_kernel<<<dim3(T_SEQ / 128, N_QH, B_SZ), 256, 0, stream>>>(Qb, Kb, Vb, attn);

  // 5) output projection -> f32 d_out
  gemm128<0><<<dim3(D_MODEL / 128, (int)MT / 128), 256, 0, stream>>>(
      attn, wot, out, nullptr, (int)MT, D_MODEL, D_MODEL, 0);
}
